// HeterogeneousGraphLayer_33500744909282
// MI455X (gfx1250) — matrix operand order repaired and checked by reading, not compiled, not run on hardware
//
#include <hip/hip_runtime.h>
#include <hip/hip_bf16.h>

typedef __bf16 bf16;
typedef __attribute__((ext_vector_type(16))) __bf16 v16bf;
typedef __attribute__((ext_vector_type(8)))  float  v8f;

#define Bb 8
#define Nn 1024
#define Hh 128
#define Rr 8

// ---------------------------------------------------------------------------
// WMMA fragment helpers (wave32, v_wmma_f32_16x16x32_bf16)
// ---------------------------------------------------------------------------

// A fragment from row-major [M][K] bf16 tile (ld in halves).
// ISA 16-bit A 16x32: lanes 0-15 row M=lane, elems 0..7 -> K 0..7, 8..15 -> K 16..23
//                     lanes 16-31 same rows, K offset +8.
__device__ __forceinline__ v16bf frag_A(const bf16* base, int ld) {
  const int lane = threadIdx.x & 31;
  const int row  = lane & 15;
  const int g    = lane >> 4;
  const bf16* p = base + row * ld + g * 8;
  v16bf a;
#pragma unroll
  for (int e = 0; e < 8; ++e) { a[e] = p[e]; a[8 + e] = p[16 + e]; }
  return a;
}

// B fragment where the tile is stored row-major [N][K] (e.g. torch weights
// [out][in]); logical B[k][n] = tile[n][k].  Lane = column n, lane group g:
// elems 0..7 -> K 8g..8g+7, elems 8..15 -> K 16+8g..16+8g+7 (matches frag_A).
__device__ __forceinline__ v16bf frag_BT(const bf16* base, int ld) {
  const int lane = threadIdx.x & 31;
  const int n    = lane & 15;
  const int g    = lane >> 4;
  const bf16* p = base + n * ld + g * 8;
  v16bf b;
#pragma unroll
  for (int e = 0; e < 8; ++e) { b[e] = p[e]; b[8 + e] = p[16 + e]; }
  return b;
}

__device__ __forceinline__ v8f wmma_bf16(v16bf a, v16bf b, v8f c) {
  return __builtin_amdgcn_wmma_f32_16x16x32_bf16(false, a, false, b,
                                                 (short)0, c, false, false);
}

__device__ __forceinline__ void zero8(v8f* acc, int n) {
  for (int j = 0; j < n; ++j)
#pragma unroll
    for (int e = 0; e < 8; ++e) acc[j][e] = 0.f;
}

// ---------------------------------------------------------------------------
// Phase 1: fused  msgs = (adj[r,b] @ X[b]) @ W_rel[r]^T + b_rel[r]
// grid (N/64, B, R); block 256 (8 waves); output x_bf[bn][r][h] bf16
// ---------------------------------------------------------------------------
__global__ __launch_bounds__(256) void k_phase1(
    const float* __restrict__ adj, const float* __restrict__ xin,
    const float* __restrict__ W_rel, const float* __restrict__ b_rel,
    bf16* __restrict__ x_bf)
{
  __shared__ bf16 As[64 * 34];    // adj tile [row][k], pad 34 (17 banks, coprime)
  __shared__ bf16 Xt[Hh * 34];    // X tile transposed [h][m]
  __shared__ bf16 Wl[Hh * 130];   // W_rel[r] as [k_out][h]
  __shared__ bf16 Tl[64 * 130];   // GEMM1 result as bf16 A operand [row][h]

  const int n0   = blockIdx.x * 64;
  const int b    = blockIdx.y;
  const int r    = blockIdx.z;
  const int t    = threadIdx.x;
  const int w    = t >> 5;
  const int lane = t & 31;
  const int rt   = w & 3;          // row tile 0..3
  const int cs   = (w >> 2) * 4;   // col tile base 0 or 4

  const float* adjp = adj + (((size_t)r * Bb + b) * Nn + n0) * Nn;
  const float* xp   = xin + (size_t)b * Nn * Hh;
  const float* wp   = W_rel + (size_t)r * Hh * Hh;

  for (int i = t; i < Hh * Hh; i += 256)
    Wl[(i >> 7) * 130 + (i & 127)] = (bf16)wp[i];

  v8f acc[4];
  zero8(acc, 4);

  for (int m0 = 0; m0 < Nn; m0 += 32) {
    // stage adj 64x32 -> bf16
    {
      const int row = t >> 2;
      const int k   = (t & 3) * 8;
      const float* src = adjp + (size_t)row * Nn + m0 + k;
      float4 f0 = *(const float4*)(src);
      float4 f1 = *(const float4*)(src + 4);
      bf16* d = &As[row * 34 + k];
      d[0] = (bf16)f0.x; d[1] = (bf16)f0.y; d[2] = (bf16)f0.z; d[3] = (bf16)f0.w;
      d[4] = (bf16)f1.x; d[5] = (bf16)f1.y; d[6] = (bf16)f1.z; d[7] = (bf16)f1.w;
      if (m0 + 32 < Nn) __builtin_prefetch(src + 32, 0, 0);
    }
    // stage X 32x128 -> transposed bf16 Xt[h][m]
    {
      const int m  = t >> 3;
      const int hb = (t & 7) * 16;
      const float* src = xp + (size_t)(m0 + m) * Hh + hb;
#pragma unroll
      for (int i = 0; i < 16; i += 4) {
        float4 f = *(const float4*)(src + i);
        Xt[(hb + i + 0) * 34 + m] = (bf16)f.x;
        Xt[(hb + i + 1) * 34 + m] = (bf16)f.y;
        Xt[(hb + i + 2) * 34 + m] = (bf16)f.z;
        Xt[(hb + i + 3) * 34 + m] = (bf16)f.w;
      }
    }
    __syncthreads();
    v16bf a = frag_A(&As[(rt * 16) * 34], 34);
#pragma unroll
    for (int j = 0; j < 4; ++j) {
      v16bf bfr = frag_BT(&Xt[((cs + j) * 16) * 34], 34);
      acc[j] = wmma_bf16(a, bfr, acc[j]);
    }
    __syncthreads();
  }

  // spill GEMM1 result (64x128) as bf16 A operand
  {
    const int g = lane >> 4;
#pragma unroll
    for (int j = 0; j < 4; ++j) {
      const int col = (cs + j) * 16 + (lane & 15);
#pragma unroll
      for (int v = 0; v < 8; ++v)
        Tl[(rt * 16 + v + 8 * g) * 130 + col] = (bf16)acc[j][v];
    }
  }
  __syncthreads();

  // GEMM2: out = Tl(64x128) @ W_rel^T, K=128
  v8f acc2[4];
  zero8(acc2, 4);
#pragma unroll
  for (int kk = 0; kk < 4; ++kk) {
    v16bf a2 = frag_A(&Tl[(rt * 16) * 130 + kk * 32], 130);
#pragma unroll
    for (int j = 0; j < 4; ++j) {
      v16bf b2 = frag_BT(&Wl[((cs + j) * 16) * 130 + kk * 32], 130);
      acc2[j] = wmma_bf16(a2, b2, acc2[j]);
    }
  }

  // epilogue: +b_rel, store node-major bf16
  {
    const int g = lane >> 4;
#pragma unroll
    for (int j = 0; j < 4; ++j) {
      const int col  = (cs + j) * 16 + (lane & 15);
      const float bi = b_rel[r * Hh + col];
#pragma unroll
      for (int v = 0; v < 8; ++v) {
        const int row   = rt * 16 + v + 8 * g;
        const size_t bn = (size_t)b * Nn + n0 + row;
        x_bf[(bn * Rr + r) * Hh + col] = (bf16)(acc2[j][v] + bi);
      }
    }
  }
}

// ---------------------------------------------------------------------------
// Phase 2a: qkv = x @ in_proj_w^T + in_proj_b   (65536 x 384 x 128)
// grid (512, 3); block 256
// ---------------------------------------------------------------------------
__global__ __launch_bounds__(256) void k_qkv(
    const bf16* __restrict__ x_bf, const float* __restrict__ w_in,
    const float* __restrict__ b_in, bf16* __restrict__ qkv)
{
  __shared__ bf16 Wl[128 * 130];
  const int t = threadIdx.x, w = t >> 5, lane = t & 31;
  const int row0 = blockIdx.x * 128;
  const int c0   = blockIdx.y * 128;

  for (int i = t; i < 128 * 128; i += 256)
    Wl[(i >> 7) * 130 + (i & 127)] =
        (bf16)w_in[(size_t)(c0 + (i >> 7)) * Hh + (i & 127)];

  v8f acc[8];
  zero8(acc, 8);
  __syncthreads();

#pragma unroll
  for (int k0 = 0; k0 < 128; k0 += 32) {
    v16bf a = frag_A(x_bf + (size_t)(row0 + w * 16) * Hh + k0, Hh);
#pragma unroll
    for (int j = 0; j < 8; ++j) {
      v16bf bfr = frag_BT(&Wl[(j * 16) * 130 + k0], 130);
      acc[j] = wmma_bf16(a, bfr, acc[j]);
    }
  }

  const int g = lane >> 4;
#pragma unroll
  for (int j = 0; j < 8; ++j) {
    const int cl   = j * 16 + (lane & 15);
    const float bi = b_in[c0 + cl];
#pragma unroll
    for (int v = 0; v < 8; ++v) {
      const int row = row0 + w * 16 + v + 8 * g;
      qkv[(size_t)row * 384 + c0 + cl] = (bf16)(acc[j][v] + bi);
    }
  }
}

// ---------------------------------------------------------------------------
// Phase 2b: per-node 4-head attention over R=8 tokens (fp32 VALU).
// grid 1024; block 256 = 8 nodes, one wave per node; lane = (head, query).
// ---------------------------------------------------------------------------
__global__ __launch_bounds__(256) void k_attn(
    const bf16* __restrict__ qkv, bf16* __restrict__ att)
{
  __shared__ bf16 qs[8 * 8 * 384];
  const int t = threadIdx.x, w = t >> 5, lane = t & 31;
  const int node = blockIdx.x * 8 + w;

  const bf16* src = qkv + (size_t)node * 8 * 384;
  bf16* base = &qs[w * 8 * 384];
  for (int c = lane; c < 384; c += 32)
    *(uint4*)(base + c * 8) = *(const uint4*)(src + c * 8);
  __syncthreads();

  const int hd = lane >> 3, qi = lane & 7;
  const bf16* qp = base + qi * 384 + hd * 32;
  float q[32];
#pragma unroll
  for (int d = 0; d < 32; ++d) q[d] = (float)qp[d];

  const float scale = 0.17677669529663687f;  // 1/sqrt(32)
  float s[8];
#pragma unroll
  for (int kj = 0; kj < 8; ++kj) {
    const bf16* kp = base + kj * 384 + 128 + hd * 32;
    float d0 = 0.f;
#pragma unroll
    for (int d = 0; d < 32; ++d) d0 += q[d] * (float)kp[d];
    s[kj] = d0 * scale;
  }
  float mx = s[0];
#pragma unroll
  for (int kj = 1; kj < 8; ++kj) mx = fmaxf(mx, s[kj]);
  float den = 0.f;
#pragma unroll
  for (int kj = 0; kj < 8; ++kj) { s[kj] = __expf(s[kj] - mx); den += s[kj]; }
  const float inv = 1.f / den;

  float o[32];
#pragma unroll
  for (int d = 0; d < 32; ++d) o[d] = 0.f;
#pragma unroll
  for (int kj = 0; kj < 8; ++kj) {
    const float p = s[kj] * inv;
    const bf16* vp = base + kj * 384 + 256 + hd * 32;
#pragma unroll
    for (int d = 0; d < 32; ++d) o[d] += p * (float)vp[d];
  }
  bf16* op = att + ((size_t)node * 8 + qi) * 128 + hd * 32;
#pragma unroll
  for (int d = 0; d < 32; ++d) op[d] = (bf16)o[d];
}

// ---------------------------------------------------------------------------
// Phase 2c: out_proj GEMM + mean over R + residual + LayerNorm, fused.
// grid 512 (16 nodes/block); block 256.  C-fragment layout => the 8 accum
// VGPRs of a lane are exactly the 8 relations of one node (node = 2w+g).
// ---------------------------------------------------------------------------
__global__ __launch_bounds__(256) void k_out(
    const bf16* __restrict__ att, const float* __restrict__ w_out,
    const float* __restrict__ b_out, const float* __restrict__ xin,
    const float* __restrict__ gamma, const float* __restrict__ beta,
    float* __restrict__ out)
{
  __shared__ bf16  Wl[128 * 130];
  __shared__ float Ys[16 * 130];
  __shared__ float ps[256], pq[256];
  __shared__ float mu_s[16], rs_s[16];

  const int t = threadIdx.x, w = t >> 5, lane = t & 31;
  const int node0 = blockIdx.x * 16;

  for (int i = t; i < 128 * 128; i += 256)
    Wl[(i >> 7) * 130 + (i & 127)] = (bf16)w_out[i];

  v8f acc[8];
  zero8(acc, 8);
  __syncthreads();

  const size_t rowbase = (size_t)node0 * 8 + w * 16;
#pragma unroll
  for (int k0 = 0; k0 < 128; k0 += 32) {
    v16bf a = frag_A(att + rowbase * 128 + k0, 128);
#pragma unroll
    for (int j = 0; j < 8; ++j) {
      v16bf bfr = frag_BT(&Wl[(j * 16) * 130 + k0], 130);
      acc[j] = wmma_bf16(a, bfr, acc[j]);
    }
  }

  const int g   = lane >> 4;
  const int ndl = 2 * w + g;        // local node 0..15
  const int bn  = node0 + ndl;
#pragma unroll
  for (int j = 0; j < 8; ++j) {
    const int cl = j * 16 + (lane & 15);
    float ssum = 0.f;
#pragma unroll
    for (int v = 0; v < 8; ++v) ssum += acc[j][v];
    const float mean = ssum * 0.125f + b_out[cl];
    Ys[ndl * 130 + cl] = xin[(size_t)bn * Hh + cl] + mean;
  }
  __syncthreads();

  {
    const int nd = t >> 4, seg = t & 15;
    float s = 0.f, q2 = 0.f;
#pragma unroll
    for (int i = 0; i < 8; ++i) {
      const float yv = Ys[nd * 130 + seg * 8 + i];
      s += yv; q2 += yv * yv;
    }
    ps[t] = s; pq[t] = q2;
  }
  __syncthreads();
  if (t < 16) {
    float s = 0.f, q2 = 0.f;
    for (int i = 0; i < 16; ++i) { s += ps[t * 16 + i]; q2 += pq[t * 16 + i]; }
    const float mu  = s * (1.f / 128.f);
    const float var = q2 * (1.f / 128.f) - mu * mu;
    mu_s[t] = mu;
    rs_s[t] = rsqrtf(var + 1e-5f);
  }
  __syncthreads();
  {
    const int nd = t >> 4, seg = t & 15;
    const float mu = mu_s[nd], rs = rs_s[nd];
    const int bn2 = node0 + nd;
#pragma unroll
    for (int i = 0; i < 8; ++i) {
      const int col = seg * 8 + i;
      out[(size_t)bn2 * Hh + col] =
          (Ys[nd * 130 + col] - mu) * rs * gamma[col] + beta[col];
    }
  }
}

// ---------------------------------------------------------------------------
extern "C" void kernel_launch(void* const* d_in, const int* in_sizes, int n_in,
                              void* d_out, int out_size, void* d_ws, size_t ws_size,
                              hipStream_t stream) {
  (void)in_sizes; (void)n_in; (void)out_size; (void)ws_size;
  const float* node_features = (const float*)d_in[0];
  const float* adj   = (const float*)d_in[1];
  const float* W_rel = (const float*)d_in[2];
  const float* b_rel = (const float*)d_in[3];
  const float* in_w  = (const float*)d_in[4];
  const float* in_b  = (const float*)d_in[5];
  const float* out_w = (const float*)d_in[6];
  const float* out_b = (const float*)d_in[7];
  const float* gamma = (const float*)d_in[8];
  const float* beta  = (const float*)d_in[9];
  float* out = (float*)d_out;

  char* ws = (char*)d_ws;
  bf16* x_bf = (bf16*)(ws);                               // 16 MB
  bf16* qkv  = (bf16*)(ws + (size_t)16777216);            // 48 MB
  bf16* att  = (bf16*)(ws + (size_t)16777216 + 50331648); // 16 MB

  k_phase1<<<dim3(Nn / 64, Bb, Rr), 256, 0, stream>>>(adj, node_features,
                                                      W_rel, b_rel, x_bf);
  k_qkv<<<dim3(512, 3), 256, 0, stream>>>(x_bf, in_w, in_b, qkv);
  k_attn<<<dim3(1024), 256, 0, stream>>>(qkv, att);
  k_out<<<dim3(512), 256, 0, stream>>>(att, out_w, out_b, node_features,
                                       gamma, beta, out);
}
